// SparseAttention_67654324847336
// MI455X (gfx1250) — matrix operand order repaired and checked by reading, not compiled, not run on hardware
//
#include <hip/hip_runtime.h>
#include <hip/hip_bf16.h>

typedef __attribute__((ext_vector_type(16))) _Float16 v16h;
typedef __attribute__((ext_vector_type(8)))  _Float16 v8h;
typedef __attribute__((ext_vector_type(8)))  float    v8f;
typedef __attribute__((ext_vector_type(4)))  unsigned int u32x4;
typedef __attribute__((ext_vector_type(8)))  unsigned int u32x8;

#define HIDDEN   1024
#define NH       16
#define HD       64
#define SEQ      2048
#define BATCH    4
#define HALF_WIN 128

#define KTILE    64                 // K-chunk staged by TDM per step
#define NITER    (HIDDEN / KTILE)   // 16

#define BAND     272   // 17 key tiles of 16 cover [q0-128, q0+143]
#define BAND_PAD 288   // rounded to 9 chunks of 32 for the P*V pass
#define LDS_COLS 296   // padded LDS row stride (floats)

// ---- WMMA operand gathers: contiguous k order (k = 16*hi + e), same in A and B ----

// A-matrix 16-bit 16x32 (MxK): lane = hi*16+ln, row m = ln,
// element e -> k = kc + hi*16 + e
__device__ __forceinline__ v16h load_a_tile(const _Float16* __restrict__ row,
                                            int kc, int hi) {
  v8h lo = *(const v8h*)(row + kc + hi * 16);
  v8h hh = *(const v8h*)(row + kc + 8 + hi * 16);
  v16h a;
#pragma unroll
  for (int i = 0; i < 8; ++i) { a[i] = lo[i]; a[8 + i] = hh[i]; }
  return a;
}

// B-matrix 16-bit 32x16 (KxN): lane n = ln, element e -> k = hi*16 + e
// (contiguous 32B per lane; caller passes row pointer for column n)

// generic pointer to LDS byte offset (low 32 bits of the shared aperture addr)
__device__ __forceinline__ unsigned lds_off(const void* p) {
  return (unsigned)(unsigned long long)(size_t)p;
}

// ---- Tensor Data Mover: stage a 64(N) x 64(K) fp16 W tile into LDS ----
// D# per ISA cdna5 ch.8: group0 = {count|flags, lds_addr, global_addr, type},
// group1 = {mask/data_size, tensor dims, tile dims, dim0 stride}.
__device__ __forceinline__ void tdm_issue_w_tile(const _Float16* gsrc,
                                                 unsigned ldsaddr) {
  unsigned long long ga = (unsigned long long)(size_t)gsrc;
  u32x4 g0;
  g0[0] = 1u;                                    // count=1, no gather/restore
  g0[1] = ldsaddr;                               // lds_addr (bytes)
  g0[2] = (unsigned)ga;                          // global_addr[31:0]
  g0[3] = (unsigned)(ga >> 32) | (2u << 30);     // global_addr[56:32], type=2
  u32x8 g1;
  g1[0] = 1u << 16;                              // wg_mask=0, data_size=2B
  g1[1] = (unsigned)HIDDEN << 16;                // tensor_dim0[15:0] @ [63:48]
  g1[2] = (unsigned)HIDDEN << 16;                // dim0 hi=0, tensor_dim1 lo
  g1[3] = (unsigned)KTILE << 16;                 // dim1 hi=0, tile_dim0=64
  g1[4] = 64u;                                   // tile_dim1=64, tile_dim2=0
  g1[5] = (unsigned)HIDDEN;                      // tensor_dim0_stride=1024
  g1[6] = 0u;
  g1[7] = 0u;
  asm volatile("tensor_load_to_lds %0, %1" :: "s"(g0), "s"(g1) : "memory");
}

// Shared GEMM mainloop: per-wave 16x64 tile, W staged in LDS by TDM,
// double-buffered on TENSORcnt. acc[nt] over 4 N-subtiles.
__device__ __forceinline__ void gemm_mainloop(
    const _Float16* __restrict__ arow, const _Float16* __restrict__ W,
    int n0, int wave, int ln, int hi, _Float16* slab, v8f acc[4]) {
  if (wave == 0) tdm_issue_w_tile(W + (size_t)n0 * HIDDEN, lds_off(slab));
  for (int it = 0; it < NITER; ++it) {
    const int kc = it * KTILE;
    if (wave == 0) {
      if (it + 1 < NITER) {
        tdm_issue_w_tile(
            W + (size_t)n0 * HIDDEN + (kc + KTILE),
            lds_off(slab) + (unsigned)(((it + 1) & 1) * KTILE * 64 * 2));
        __builtin_amdgcn_s_wait_tensorcnt(1);  // DMA(it) complete (in-order)
      } else {
        __builtin_amdgcn_s_wait_tensorcnt(0);
      }
    }
    __syncthreads();  // slab[it&1] visible to all waves
    const _Float16* sw = slab + (it & 1) * (KTILE * 64);
    v16h a0 = load_a_tile(arow, kc, hi);
    v16h a1 = load_a_tile(arow, kc + 32, hi);
#pragma unroll
    for (int nt = 0; nt < 4; ++nt) {
      const _Float16* srow = sw + (nt * 16 + ln) * KTILE;
      v16h b0 = *(const v16h*)(srow + hi * 16);
      v16h b1 = *(const v16h*)(srow + 32 + hi * 16);
      acc[nt] = __builtin_amdgcn_wmma_f32_16x16x32_f16(
          false, a0, false, b0, (short)0, acc[nt], false, false);
      acc[nt] = __builtin_amdgcn_wmma_f32_16x16x32_f16(
          false, a1, false, b1, (short)0, acc[nt], false, false);
    }
    __syncthreads();  // all waves done with slab[it&1] before DMA(it+2)
  }
}

// ---------------- stage 0: fp32 -> fp16 convert ----------------
__global__ void cvt_f32_f16(const float* __restrict__ src,
                            _Float16* __restrict__ dst, int n) {
  int i = blockIdx.x * blockDim.x + threadIdx.x;
  int stride = gridDim.x * blockDim.x;
  for (; i < n; i += stride) dst[i] = (_Float16)src[i];
}

// ---------------- stage 1: fused QKV projection ----------------
// y = X @ W[which]^T + b[which];  Q additionally scaled by 1/sqrt(HD)
// Q,K stored [B*NH, S, HD] f16; V stored transposed [B*NH, HD, S] f16
__global__ __launch_bounds__(256) void qkv_gemm(
    const _Float16* __restrict__ X, const _Float16* __restrict__ Wall,
    const float* __restrict__ bq, const float* __restrict__ bk,
    const float* __restrict__ bv,
    _Float16* __restrict__ Q, _Float16* __restrict__ K,
    _Float16* __restrict__ Vt) {
  __shared__ __align__(16) _Float16 slab[2 * KTILE * 64];
  const int lane = threadIdx.x & 31, wave = threadIdx.x >> 5;
  const int ln = lane & 15, hi = lane >> 4;
  const int m0 = blockIdx.x * 128 + wave * 16;
  const int n0 = blockIdx.y * 64;
  const int which = blockIdx.z;
  const _Float16* W = Wall + (size_t)which * HIDDEN * HIDDEN;
  const float* bias = (which == 0) ? bq : ((which == 1) ? bk : bv);

  v8f acc[4] = {};
  gemm_mainloop(X + (size_t)(m0 + ln) * HIDDEN, W, n0, wave, ln, hi, slab, acc);

#pragma unroll
  for (int nt = 0; nt < 4; ++nt) {
    int n = n0 + nt * 16 + ln;
    int h = n >> 6, d = n & 63;
    float bn = bias[n];
#pragma unroll
    for (int v = 0; v < 8; ++v) {
      int m = m0 + v + 8 * hi;
      int b = m / SEQ, s = m & (SEQ - 1);
      float val = acc[nt][v] + bn;
      if (which == 0) val *= 0.125f;  // 1/sqrt(64) folded into Q
      _Float16 hv = (_Float16)val;
      size_t bh = (size_t)(b * NH + h);
      if (which == 2)      Vt[(bh * HD + d) * SEQ + s] = hv;
      else if (which == 0) Q[(bh * SEQ + s) * HD + d] = hv;
      else                 K[(bh * SEQ + s) * HD + d] = hv;
    }
  }
}

// ---------------- stage 2: banded (local-window) attention ----------------
__global__ __launch_bounds__(64) void attn_local(
    const _Float16* __restrict__ Q, const _Float16* __restrict__ K,
    const _Float16* __restrict__ Vt, _Float16* __restrict__ ctx) {
  __shared__ float sc[2][16][LDS_COLS];
  const int lane = threadIdx.x & 31, wave = threadIdx.x >> 5;
  const int ln = lane & 15, hi = lane >> 4;
  const int wid = blockIdx.x * 2 + wave;
  const int bh = wid >> 7;              // / (SEQ/16)
  const int q0 = (wid & 127) << 4;
  float (*S_)[LDS_COLS] = sc[wave];

  // Q A-tiles (head_dim 64 = two K=32 chunks), pre-scaled at projection
  const _Float16* qrow = Q + ((size_t)bh * SEQ + q0 + ln) * HD;
  v16h aq0 = load_a_tile(qrow, 0, hi);
  v16h aq1 = load_a_tile(qrow, 32, hi);

  const int kstart = q0 - HALF_WIN;

  // phase 1: scores = Q K^T over the 272-key band, window mask applied
  for (int t = 0; t < 17; ++t) {
    int base = kstart + 16 * t;       // always multiple of 16
    if (base < 0 || base >= SEQ) {    // tile fully outside sequence
#pragma unroll
      for (int v = 0; v < 8; ++v) S_[v + 8 * hi][16 * t + ln] = -1.0e30f;
    } else {
      const _Float16* krow = K + ((size_t)bh * SEQ + base + ln) * HD;
      v16h bk0 = *(const v16h*)(krow + hi * 16);
      v16h bk1 = *(const v16h*)(krow + 32 + hi * 16);
      v8f c = {};
      c = __builtin_amdgcn_wmma_f32_16x16x32_f16(false, aq0, false, bk0,
                                                 (short)0, c, false, false);
      c = __builtin_amdgcn_wmma_f32_16x16x32_f16(false, aq1, false, bk1,
                                                 (short)0, c, false, false);
      int key = base + ln;
#pragma unroll
      for (int v = 0; v < 8; ++v) {
        int m = v + 8 * hi;
        int d = (q0 + m) - key;
        float sval = c[v];
        if (d > HALF_WIN || d < -HALF_WIN) sval -= 10000.0f;  // reference mask
        S_[m][16 * t + ln] = sval;
      }
    }
  }
  __syncthreads();

  // phase 2: softmax; lane pair (ln, ln+16) splits row ln across 2x136 cols
  {
    int row = ln;
    int c0 = hi * (BAND / 2);
    float mx = -3.0e38f;
    for (int c = 0; c < BAND / 2; ++c) mx = fmaxf(mx, S_[row][c0 + c]);
    mx = fmaxf(mx, __shfl_xor(mx, 16, 32));
    float sum = 0.f;
    for (int c = 0; c < BAND / 2; ++c) {
      float e = __expf(S_[row][c0 + c] - mx);
      S_[row][c0 + c] = e;
      sum += e;
    }
    sum += __shfl_xor(sum, 16, 32);
    float inv = 1.0f / sum;
    for (int c = 0; c < BAND / 2; ++c) S_[row][c0 + c] *= inv;
#pragma unroll
    for (int c = 0; c < 8; ++c) S_[row][BAND + hi * 8 + c] = 0.f;  // pad
  }
  __syncthreads();

  // phase 3: ctx = P * V  (A = probs from LDS in A-layout, B = rows of V^T)
  v8f ca[4] = {};
  const _Float16* vbase = Vt + (size_t)bh * HD * SEQ;
  for (int t2 = 0; t2 < BAND_PAD / 32; ++t2) {
    v16h pa;
#pragma unroll
    for (int e = 0; e < 16; ++e) {
      int k = hi * 16 + e;
      pa[e] = (_Float16)S_[ln][32 * t2 + k];
    }
    int key0 = kstart + 32 * t2 + hi * 16;   // clamp: OOB cols have prob 0
    if (key0 < 0) key0 = 0;
    if (key0 > SEQ - 16) key0 = SEQ - 16;
#pragma unroll
    for (int nt = 0; nt < 4; ++nt) {
      const _Float16* vrow = vbase + (size_t)(nt * 16 + ln) * SEQ;
      v16h bv = *(const v16h*)(vrow + key0);
      ca[nt] = __builtin_amdgcn_wmma_f32_16x16x32_f16(
          false, pa, false, bv, (short)0, ca[nt], false, false);
    }
  }
  // store ctx as f16 [B, S, HIDDEN] for the output projection
  int b = bh >> 4, h = bh & 15;
#pragma unroll
  for (int nt = 0; nt < 4; ++nt) {
    int dch = h * HD + nt * 16 + ln;
#pragma unroll
    for (int v = 0; v < 8; ++v) {
      int s = q0 + v + 8 * hi;
      ctx[((size_t)b * SEQ + s) * HIDDEN + dch] = (_Float16)ca[nt][v];
    }
  }
}

// ---------------- stage 3: output projection (f32 out) ----------------
__global__ __launch_bounds__(256) void out_gemm(
    const _Float16* __restrict__ Ctx, const _Float16* __restrict__ Wo,
    const float* __restrict__ bo, float* __restrict__ out) {
  __shared__ __align__(16) _Float16 slab[2 * KTILE * 64];
  const int lane = threadIdx.x & 31, wave = threadIdx.x >> 5;
  const int ln = lane & 15, hi = lane >> 4;
  const int m0 = blockIdx.x * 128 + wave * 16;
  const int n0 = blockIdx.y * 64;

  v8f acc[4] = {};
  gemm_mainloop(Ctx + (size_t)(m0 + ln) * HIDDEN, Wo, n0, wave, ln, hi, slab,
                acc);

#pragma unroll
  for (int nt = 0; nt < 4; ++nt) {
    int n = n0 + nt * 16 + ln;
    float bn = bo[n];
#pragma unroll
    for (int v = 0; v < 8; ++v) {
      int m = m0 + v + 8 * hi;
      out[(size_t)m * HIDDEN + n] = acc[nt][v] + bn;
    }
  }
}

// ---------------- launch ----------------
extern "C" void kernel_launch(void* const* d_in, const int* in_sizes, int n_in,
                              void* d_out, int out_size, void* d_ws,
                              size_t ws_size, hipStream_t stream) {
  (void)in_sizes; (void)n_in; (void)out_size; (void)ws_size;
  const float* hs = (const float*)d_in[0];
  const float* Wq = (const float*)d_in[1];
  const float* bq = (const float*)d_in[2];
  const float* Wk = (const float*)d_in[3];
  const float* bk = (const float*)d_in[4];
  const float* Wv = (const float*)d_in[5];
  const float* bv = (const float*)d_in[6];
  const float* Wo = (const float*)d_in[7];
  const float* bo = (const float*)d_in[8];
  float* out = (float*)d_out;

  const size_t NTOK = (size_t)BATCH * SEQ;       // 8192
  const size_t XSZ = NTOK * HIDDEN;              // 8.39M elems
  const size_t WSZ = (size_t)HIDDEN * HIDDEN;    // 1.05M elems

  char* ws = (char*)d_ws;
  _Float16* X16  = (_Float16*)ws; ws += XSZ * 2;
  _Float16* W16  = (_Float16*)ws; ws += 4 * WSZ * 2;  // Wq,Wk,Wv,Wo
  _Float16* Q16  = (_Float16*)ws; ws += XSZ * 2;
  _Float16* K16  = (_Float16*)ws; ws += XSZ * 2;
  _Float16* Vt16 = (_Float16*)ws; ws += XSZ * 2;
  _Float16* C16  = (_Float16*)ws; ws += XSZ * 2;

  cvt_f32_f16<<<2048, 256, 0, stream>>>(hs, X16, (int)XSZ);
  cvt_f32_f16<<<1024, 256, 0, stream>>>(Wq, W16 + 0 * WSZ, (int)WSZ);
  cvt_f32_f16<<<1024, 256, 0, stream>>>(Wk, W16 + 1 * WSZ, (int)WSZ);
  cvt_f32_f16<<<1024, 256, 0, stream>>>(Wv, W16 + 2 * WSZ, (int)WSZ);
  cvt_f32_f16<<<1024, 256, 0, stream>>>(Wo, W16 + 3 * WSZ, (int)WSZ);

  qkv_gemm<<<dim3((unsigned)(NTOK / 128), HIDDEN / 64, 3), dim3(256), 0,
             stream>>>(X16, W16, bq, bk, bv, Q16, K16, Vt16);

  attn_local<<<dim3((BATCH * NH * (SEQ / 16)) / 2), dim3(64), 0, stream>>>(
      Q16, K16, Vt16, C16);

  out_gemm<<<dim3((unsigned)(NTOK / 128), HIDDEN / 64), dim3(256), 0,
             stream>>>(C16, W16 + 3 * WSZ, bo, out);
}
